// MultiheadAttention_2568390443414
// MI455X (gfx1250) — matrix operand order repaired and checked by reading, not compiled, not run on hardware
//
#include <hip/hip_runtime.h>

typedef __attribute__((ext_vector_type(16))) _Float16 v16h;
typedef __attribute__((ext_vector_type(8)))  _Float16 v8h;
typedef __attribute__((ext_vector_type(8)))  float    v8f;
typedef __attribute__((ext_vector_type(4)))  float    v4f;

#define BSZ 2
#define SEQ 2048
#define DMODEL 1024
#define NHEAD 16
#define DKH 64

__device__ __forceinline__ v8f wmma16(v16h a, v16h b, v8f c) {
  // v_wmma_f32_16x16x32_f16: D = A(16x32 f16) * B(32x16 f16) + C(16x16 f32)
  return __builtin_amdgcn_wmma_f32_16x16x32_f16(false, a, false, b, (short)0, c,
                                                false, false);
}

__device__ __forceinline__ v8f zero8() {
  v8f z;
#pragma unroll
  for (int j = 0; j < 8; ++j) z[j] = 0.0f;
  return z;
}

__device__ __forceinline__ v16h pack2(v8h lo, v8h hi) {
  v16h r;
#pragma unroll
  for (int j = 0; j < 8; ++j) { r[j] = lo[j]; r[8 + j] = hi[j]; }
  return r;
}

// ---------------------------------------------------------------------------
// Elementwise f32 -> f16 conversion (one pass per tensor).
// ---------------------------------------------------------------------------
__global__ __launch_bounds__(256) void cvt_f16_kernel(
    const float* __restrict__ in, _Float16* __restrict__ out, int n) {
  const int i = (blockIdx.x * 256 + threadIdx.x) * 8;
  if (i >= n) return;
  v4f a = *(const v4f*)(in + i);
  v4f b = *(const v4f*)(in + i + 4);
  v8h o;
#pragma unroll
  for (int j = 0; j < 4; ++j) {
    o[j] = (_Float16)a[j];
    o[4 + j] = (_Float16)b[j];
  }
  *(v8h*)(out + i) = o;
}

// ---------------------------------------------------------------------------
// Projection GEMM: Y = Xh[4096,1024] @ Wh^T + b  (all f16 in, f16 out).
// Wave = 32(M) x 64(N) tile; block = 8 waves = 256(M) x 64(N).
// Double-buffered K loop (unrolled x2), 8 WMMAs per 32-K step.
// vmode 0: write [B,H,S,dk] (Q,K). vmode 1: write [B,H,dk,S] (V transposed).
// ---------------------------------------------------------------------------
struct PFrag {
  v16h a0, a1, b0, b1, b2, b3;
};

__device__ __forceinline__ void load_pfrag(PFrag& f, const _Float16* __restrict__ Xh,
                                           const _Float16* __restrict__ Wh, int r0,
                                           int r1, int n0, int lr, int hf, int k0) {
  const _Float16* xa0 = Xh + (size_t)r0 * DMODEL + k0 + hf * 16;
  const _Float16* xa1 = Xh + (size_t)r1 * DMODEL + k0 + hf * 16;
  f.a0 = pack2(*(const v8h*)xa0, *(const v8h*)(xa0 + 8));
  f.a1 = pack2(*(const v8h*)xa1, *(const v8h*)(xa1 + 8));
  const int kb = k0 + hf * 16;
  f.b0 = *(const v16h*)(Wh + (size_t)(n0 + lr) * DMODEL + kb);
  f.b1 = *(const v16h*)(Wh + (size_t)(n0 + 16 + lr) * DMODEL + kb);
  f.b2 = *(const v16h*)(Wh + (size_t)(n0 + 32 + lr) * DMODEL + kb);
  f.b3 = *(const v16h*)(Wh + (size_t)(n0 + 48 + lr) * DMODEL + kb);
}

__device__ __forceinline__ void pfrag_wmma(const PFrag& f, v8f acc[2][4]) {
  acc[0][0] = wmma16(f.a0, f.b0, acc[0][0]);
  acc[0][1] = wmma16(f.a0, f.b1, acc[0][1]);
  acc[0][2] = wmma16(f.a0, f.b2, acc[0][2]);
  acc[0][3] = wmma16(f.a0, f.b3, acc[0][3]);
  acc[1][0] = wmma16(f.a1, f.b0, acc[1][0]);
  acc[1][1] = wmma16(f.a1, f.b1, acc[1][1]);
  acc[1][2] = wmma16(f.a1, f.b2, acc[1][2]);
  acc[1][3] = wmma16(f.a1, f.b3, acc[1][3]);
}

__global__ __launch_bounds__(256) void proj_kernel(
    const _Float16* __restrict__ Xh, const _Float16* __restrict__ Wh,
    const float* __restrict__ bias, _Float16* __restrict__ out, int vmode) {
  const int lane = threadIdx.x & 31, wave = threadIdx.x >> 5;
  const int lr = lane & 15, hf = lane >> 4;
  const int m0 = blockIdx.x * 256 + wave * 32;
  const int n0 = blockIdx.y * 64;
  const int r0 = m0 + lr, r1 = m0 + 16 + lr;

  v8f acc[2][4];
#pragma unroll
  for (int mt = 0; mt < 2; ++mt)
#pragma unroll
    for (int t = 0; t < 4; ++t) acc[mt][t] = zero8();

  PFrag cur, nxt;
  load_pfrag(cur, Xh, Wh, r0, r1, n0, lr, hf, 0);
  for (int k0 = 0; k0 < DMODEL; k0 += 64) {
    load_pfrag(nxt, Xh, Wh, r0, r1, n0, lr, hf, k0 + 32);
    pfrag_wmma(cur, acc);
    if (k0 + 64 < DMODEL) load_pfrag(cur, Xh, Wh, r0, r1, n0, lr, hf, k0 + 64);
    pfrag_wmma(nxt, acc);
  }

#pragma unroll
  for (int mt = 0; mt < 2; ++mt)
#pragma unroll
    for (int t = 0; t < 4; ++t) {
      const int n = n0 + t * 16 + lr;
      const int h = n >> 6, d = n & 63;
      const float bv = bias[n];
#pragma unroll
      for (int g = 0; g < 8; ++g) {
        const int m = m0 + mt * 16 + g + hf * 8;
        const int bb = m >> 11, s = m & (SEQ - 1);
        const float val = acc[mt][t][g] + bv;
        if (vmode == 0)
          out[(((size_t)(bb * NHEAD + h) * SEQ) + s) * DKH + d] = (_Float16)val;
        else
          out[(((size_t)(bb * NHEAD + h) * DKH) + d) * SEQ + s] = (_Float16)val;
      }
    }
}

// ---------------------------------------------------------------------------
// Flash attention. One wave per 16-query tile of one (b,h).
// Per 32-key chunk: all K/V fragments loaded up front (overlaps with the
// 4 score WMMAs + softmax VALU), then 4 PV WMMAs. Next chunk prefetched.
// ---------------------------------------------------------------------------
__global__ __launch_bounds__(128) void attn_kernel(
    const _Float16* __restrict__ qh, const _Float16* __restrict__ kh,
    const _Float16* __restrict__ vht, const int* __restrict__ mask,
    _Float16* __restrict__ ctx) {
  __shared__ _Float16 sP[4][16][32];  // per-wave 16x32 prob tile (4 KB)
  const int lane = threadIdx.x & 31, wave = threadIdx.x >> 5;
  const int lr = lane & 15, hf = lane >> 4;
  const int b = blockIdx.z, h = blockIdx.y;
  const int q0 = blockIdx.x * 64 + wave * 16;

  const _Float16* Qp = qh + (size_t)(b * NHEAD + h) * SEQ * DKH;
  const _Float16* Kp = kh + (size_t)(b * NHEAD + h) * SEQ * DKH;
  const _Float16* Vp = vht + (size_t)(b * NHEAD + h) * DKH * SEQ;

  // Preload Q fragments for dk=64 (two 32-wide K-steps).
  const int qrow = q0 + lr;
  v16h Aq[2];
#pragma unroll
  for (int ks = 0; ks < 2; ++ks) {
    const _Float16* p = Qp + (size_t)qrow * DKH + ks * 32 + hf * 16;
    Aq[ks] = pack2(*(const v8h*)p, *(const v8h*)(p + 8));
  }

  float mrow[8], lrow[8];
#pragma unroll
  for (int g = 0; g < 8; ++g) { mrow[g] = -3.0e38f; lrow[g] = 0.0f; }
  v8f acc[4];
#pragma unroll
  for (int t = 0; t < 4; ++t) acc[t] = zero8();

  const float scale = 0.125f;  // 1/sqrt(64)

  for (int kk = 0; kk < SEQ; kk += 32) {
    // ---- issue all chunk loads up front --------------------------------
    const _Float16* kp0 = Kp + (size_t)(kk + lr) * DKH + hf * 16;
    const _Float16* kp1 = Kp + (size_t)(kk + 16 + lr) * DKH + hf * 16;
    const v16h K00 = *(const v16h*)kp0;
    const v16h K01 = *(const v16h*)(kp0 + 32);
    const v16h K10 = *(const v16h*)kp1;
    const v16h K11 = *(const v16h*)(kp1 + 32);
    v16h Vf[4];
#pragma unroll
    for (int t = 0; t < 4; ++t)
      Vf[t] = *(const v16h*)(Vp + (size_t)(t * 16 + lr) * SEQ + kk + hf * 16);
    if (kk + 32 < SEQ) {
      __builtin_prefetch(Kp + (size_t)(kk + 32 + lr) * DKH, 0, 1);
      __builtin_prefetch(Vp + (size_t)lr * SEQ + kk + 32, 0, 1);
    }
    // ---- scores: S = Q K^T ---------------------------------------------
    v8f sc[2];
    sc[0] = zero8();
    sc[0] = wmma16(Aq[0], K00, sc[0]);
    sc[0] = wmma16(Aq[1], K01, sc[0]);
    sc[1] = zero8();
    sc[1] = wmma16(Aq[0], K10, sc[1]);
    sc[1] = wmma16(Aq[1], K11, sc[1]);
    // ---- mask + online softmax (row = 16 lanes of one half-wave) -------
#pragma unroll
    for (int g = 0; g < 8; ++g) {
      const int m = q0 + g + hf * 8;
      float s0 = sc[0][g] * scale, s1 = sc[1][g] * scale;
      const int* mp = mask + (size_t)(b * SEQ + m) * SEQ + kk + lr;
      if (mp[0] == 0)  s0 = -10000.0f;
      if (mp[16] == 0) s1 = -10000.0f;
      float mx = fmaxf(s0, s1);
      mx = fmaxf(mx, __shfl_xor(mx, 1, 32));
      mx = fmaxf(mx, __shfl_xor(mx, 2, 32));
      mx = fmaxf(mx, __shfl_xor(mx, 4, 32));
      mx = fmaxf(mx, __shfl_xor(mx, 8, 32));
      const float mnew = fmaxf(mrow[g], mx);
      const float corr = __expf(mrow[g] - mnew);
      const float p0 = __expf(s0 - mnew);
      const float p1 = __expf(s1 - mnew);
      float ps = p0 + p1;
      ps += __shfl_xor(ps, 1, 32);
      ps += __shfl_xor(ps, 2, 32);
      ps += __shfl_xor(ps, 4, 32);
      ps += __shfl_xor(ps, 8, 32);
      lrow[g] = lrow[g] * corr + ps;
      mrow[g] = mnew;
#pragma unroll
      for (int t = 0; t < 4; ++t) acc[t][g] *= corr;
      sP[wave][g + hf * 8][lr]      = (_Float16)p0;
      sP[wave][g + hf * 8][16 + lr] = (_Float16)p1;
    }
    // D-layout -> A-layout through per-wave LDS (in-wave ordering).
    asm volatile("s_wait_dscnt 0" ::: "memory");
    const v8h pr0 = *(const v8h*)&sP[wave][lr][hf * 16];
    const v8h pr1 = *(const v8h*)&sP[wave][lr][hf * 16 + 8];
    asm volatile("" ::: "memory");
    const v16h Ap = pack2(pr0, pr1);
    // ---- PV: out += P V -------------------------------------------------
#pragma unroll
    for (int t = 0; t < 4; ++t) acc[t] = wmma16(Ap, Vf[t], acc[t]);
  }

  // ---- normalize + write ctx [B,S,H*dk] as f16 --------------------------
#pragma unroll
  for (int t = 0; t < 4; ++t) {
    const int d = t * 16 + lr;
#pragma unroll
    for (int g = 0; g < 8; ++g) {
      const int q = q0 + g + hf * 8;
      const float o = acc[t][g] / lrow[g];
      ctx[(size_t)(b * SEQ + q) * DMODEL + h * DKH + d] = (_Float16)o;
    }
  }
}

// ---------------------------------------------------------------------------
// Output projection: out = ctx[4096,1024](f16) @ Who^T + bo   (f32 out)
// Same 32x64-per-wave double-buffered scheme as proj_kernel.
// ---------------------------------------------------------------------------
__global__ __launch_bounds__(256) void oproj_kernel(
    const _Float16* __restrict__ ctx, const _Float16* __restrict__ Wh,
    const float* __restrict__ bias, float* __restrict__ out) {
  const int lane = threadIdx.x & 31, wave = threadIdx.x >> 5;
  const int lr = lane & 15, hf = lane >> 4;
  const int m0 = blockIdx.x * 256 + wave * 32;
  const int n0 = blockIdx.y * 64;
  const int r0 = m0 + lr, r1 = m0 + 16 + lr;

  v8f acc[2][4];
#pragma unroll
  for (int mt = 0; mt < 2; ++mt)
#pragma unroll
    for (int t = 0; t < 4; ++t) acc[mt][t] = zero8();

  PFrag cur, nxt;
  load_pfrag(cur, ctx, Wh, r0, r1, n0, lr, hf, 0);
  for (int k0 = 0; k0 < DMODEL; k0 += 64) {
    load_pfrag(nxt, ctx, Wh, r0, r1, n0, lr, hf, k0 + 32);
    pfrag_wmma(cur, acc);
    if (k0 + 64 < DMODEL) load_pfrag(cur, ctx, Wh, r0, r1, n0, lr, hf, k0 + 64);
    pfrag_wmma(nxt, acc);
  }

#pragma unroll
  for (int mt = 0; mt < 2; ++mt)
#pragma unroll
    for (int t = 0; t < 4; ++t) {
      const int n = n0 + t * 16 + lr;
      const float bv = bias[n];
#pragma unroll
      for (int g = 0; g < 8; ++g) {
        const int m = m0 + mt * 16 + g + hf * 8;
        out[(size_t)m * DMODEL + n] = acc[mt][t][g] + bv;
      }
    }
}

extern "C" void kernel_launch(void* const* d_in, const int* in_sizes, int n_in,
                              void* d_out, int out_size, void* d_ws,
                              size_t ws_size, hipStream_t stream) {
  const float* q    = (const float*)d_in[0];
  const float* k    = (const float*)d_in[1];
  const float* v    = (const float*)d_in[2];
  const int*   mask = (const int*)d_in[3];
  const float* Wq = (const float*)d_in[4],  *bq = (const float*)d_in[5];
  const float* Wk = (const float*)d_in[6],  *bk = (const float*)d_in[7];
  const float* Wv = (const float*)d_in[8],  *bv = (const float*)d_in[9];
  const float* Wo = (const float*)d_in[10], *bo = (const float*)d_in[11];

  const size_t NX = (size_t)BSZ * SEQ * DMODEL;     // 4,194,304
  const size_t NW = (size_t)DMODEL * DMODEL;        // 1,048,576
  _Float16* xq  = (_Float16*)d_ws;
  _Float16* xk  = xq + NX;
  _Float16* xv  = xk + NX;
  _Float16* whq = xv + NX;
  _Float16* whk = whq + NW;
  _Float16* whv = whk + NW;
  _Float16* who = whv + NW;
  _Float16* qh  = who + NW;
  _Float16* kh  = qh + NX;
  _Float16* vht = kh + NX;
  _Float16* ctx = vht + NX;

  // f32 -> f16 staging (one pass each)
  cvt_f16_kernel<<<NX / (256 * 8), 256, 0, stream>>>(q, xq, (int)NX);
  cvt_f16_kernel<<<NX / (256 * 8), 256, 0, stream>>>(k, xk, (int)NX);
  cvt_f16_kernel<<<NX / (256 * 8), 256, 0, stream>>>(v, xv, (int)NX);
  cvt_f16_kernel<<<NW / (256 * 8), 256, 0, stream>>>(Wq, whq, (int)NW);
  cvt_f16_kernel<<<NW / (256 * 8), 256, 0, stream>>>(Wk, whk, (int)NW);
  cvt_f16_kernel<<<NW / (256 * 8), 256, 0, stream>>>(Wv, whv, (int)NW);
  cvt_f16_kernel<<<NW / (256 * 8), 256, 0, stream>>>(Wo, who, (int)NW);

  dim3 pgrid(4096 / 256, 1024 / 64, 1);
  proj_kernel<<<pgrid, 256, 0, stream>>>(xq, whq, bq, qh, 0);
  proj_kernel<<<pgrid, 256, 0, stream>>>(xk, whk, bk, kh, 0);
  proj_kernel<<<pgrid, 256, 0, stream>>>(xv, whv, bv, vht, 1);

  dim3 agrid(SEQ / 64, NHEAD, BSZ);
  attn_kernel<<<agrid, 128, 0, stream>>>(qh, kh, vht, mask, ctx);

  oproj_kernel<<<pgrid, 256, 0, stream>>>(ctx, who, bo, (float*)d_out);
}
